// Attention2D_ROPE_67233418051753
// MI455X (gfx1250) — matrix operand order repaired and checked by reading, not compiled, not run on hardware
//
#include <hip/hip_runtime.h>

// ---------------------------------------------------------------------------
// 2D attention + RoPE for MI455X (gfx1250, wave32, WMMA + TDM).
// Pipeline: cvt(f32->bf16) -> QKV GEMM (wmma bf16) -> RoPE(q,k in place)
//           -> flash-style attention (wmma bf16, S^T/O^T trick)
//           -> out GEMM (wmma bf16, f32 result + bias).
// All matmuls accumulate in f32 via v_wmma_f32_16x16x32_bf16.
// LDS tile staging uses the Tensor Data Mover (tensor_load_to_lds +
// s_wait_tensorcnt) when the toolchain exposes it; the TDM pad feature
// reproduces the bank-spreading LDS strides (72 / 40 elements).
// ---------------------------------------------------------------------------

typedef __bf16 bf16;
typedef __attribute__((ext_vector_type(16))) bf16  v16bf;
typedef __attribute__((ext_vector_type(8)))  bf16  v8bf;
typedef __attribute__((ext_vector_type(4)))  bf16  v4bf;
typedef __attribute__((ext_vector_type(8)))  float v8f;
typedef __attribute__((ext_vector_type(4)))  unsigned int v4u;
typedef __attribute__((ext_vector_type(8)))  int v8i;
typedef __attribute__((ext_vector_type(4)))  int v4i;

#define WMMA_BF16(a, b, c) \
  __builtin_amdgcn_wmma_f32_16x16x32_bf16(false, (a), false, (b), (short)0, (c), false, false)

#if defined(__gfx1250__) && __has_builtin(__builtin_amdgcn_tensor_load_to_lds) && \
    __has_builtin(__builtin_amdgcn_s_wait_tensorcnt)
#define USE_TDM 1
#else
#define USE_TDM 0
#endif

#if USE_TDM
// Issue one TDM 2D tile load: tile_h rows of tile_w bf16 elements, global row
// stride row_stride elements, into LDS at lds_addr with (pad_int, pad_amt)
// LDS padding codes (interval = 2^(pad_int+1) DWORDs, amount = pad_amt+1 DWORDs).
__device__ __forceinline__ void tdm_load_2d(unsigned lds_addr, const bf16* gptr,
                                            unsigned tile_w, unsigned tile_h,
                                            unsigned long long row_stride,
                                            unsigned pad_int, unsigned pad_amt) {
  unsigned long long ga = (unsigned long long)(size_t)gptr;
  v4u g0 = { 1u,                                             // count=1 (valid)
             lds_addr,                                       // lds_addr
             (unsigned)(ga & 0xFFFFFFFFu),                   // global_addr lo
             (unsigned)((ga >> 32) & 0x01FFFFFFu) | (2u << 30) };  // ga hi | type=2
  v8i g1 = {
    (int)((1u << 16)                 // data_size = 2 bytes (bf16)
          | (1u << 20)               // pad_enable
          | (pad_int << 22) | (pad_amt << 25)),
    (int)((tile_w & 0xFFFFu) << 16),                        // tensor_dim0 lo16
    (int)((tile_w >> 16) | ((tile_h & 0xFFFFu) << 16)),     // td0 hi | td1 lo
    (int)((tile_h >> 16) | (tile_w << 16)),                 // td1 hi | tile_dim0
    (int)(tile_h),                                          // tile_dim1 (tile_dim2=0)
    (int)(row_stride & 0xFFFFFFFFull),                      // dim0 stride lo32
    (int)((row_stride >> 32) & 0xFFFFull),                  // dim0 stride hi16
    0 };
  v4i gz = { 0, 0, 0, 0 };
#if __has_include(<hip/amd_detail/amd_gfx1250_TDM.h>)
  v8i gx = { 0, 0, 0, 0, 0, 0, 0, 0 };
  __builtin_amdgcn_tensor_load_to_lds(g0, g1, gz, gz, gx, 0);   // 6-arg toolchain
#else
  __builtin_amdgcn_tensor_load_to_lds(g0, g1, gz, gz, 0);       // 5-arg (ROCm 7.2)
#endif
}
#endif

// Combine two 8-element bf16 chunks (two b128 loads) into one 16-elem frag.
__device__ __forceinline__ v16bf ld_frag16(const bf16* lo_p, const bf16* hi_p) {
  v8bf lo = *(const v8bf*)lo_p;
  v8bf hi = *(const v8bf*)hi_p;
  return __builtin_shufflevector(lo, hi, 0,1,2,3,4,5,6,7,8,9,10,11,12,13,14,15);
}

// ---------------------------------------------------------------------------
// f32 -> bf16 convert, 4 elements/thread
// ---------------------------------------------------------------------------
__global__ void cvt_kernel(const float* __restrict__ in, bf16* __restrict__ out, int n4) {
  int i = blockIdx.x * blockDim.x + threadIdx.x;
  if (i < n4) {
    float4 v = ((const float4*)in)[i];
    v4bf o;
    o[0] = (bf16)v.x; o[1] = (bf16)v.y; o[2] = (bf16)v.z; o[3] = (bf16)v.w;
    ((v4bf*)out)[i] = o;
  }
}

// ---------------------------------------------------------------------------
// Tiled WMMA GEMM: C[M x N] = A[M x K](bf16,row-major) * W[N x K]^T(bf16) + bias
// Block tile 128x128, BK=32, 256 threads = 8 waves; wave tile 32x64.
// A/B tiles staged into LDS by TDM (pad 4 DWORDs per 16-DWORD row -> stride 40).
// ---------------------------------------------------------------------------
template <bool OUT_F32>
__global__ __launch_bounds__(256) void gemm_kernel(
    const bf16* __restrict__ A, const bf16* __restrict__ W,
    const float* __restrict__ bias, float* __restrict__ Cf, bf16* __restrict__ Cb,
    int M, int Nn, int K) {
  __shared__ bf16 lA[128 * 40];  // padded stride 40 elems (80B, 16B aligned)
  __shared__ bf16 lB[128 * 40];

  const int tid  = threadIdx.x;
  const int lane = tid & 31;
  const int wv   = tid >> 5;
  const int r    = lane & 15;
  const int hh   = lane >> 4;          // lane half
  const int wm   = (wv & 3) * 32;      // wave m offset within block
  const int wn   = (wv >> 2) * 64;     // wave n offset within block
  const int mb   = blockIdx.y * 128;
  const int nb   = blockIdx.x * 128;

  v8f acc[2][4] = {};

  for (int kb = 0; kb < K; kb += 32) {
#if USE_TDM
    // --- TDM stages both tiles; one descriptor per matrix, waves 0 and 1 ---
    if (wv == 0) {
      tdm_load_2d((unsigned)(size_t)&lA[0], A + (size_t)mb * K + kb,
                  32u, 128u, (unsigned long long)K, 3u, 3u);
      __builtin_amdgcn_s_wait_tensorcnt(0);
    } else if (wv == 1) {
      tdm_load_2d((unsigned)(size_t)&lB[0], W + (size_t)nb * K + kb,
                  32u, 128u, (unsigned long long)K, 3u, 3u);
      __builtin_amdgcn_s_wait_tensorcnt(0);
    }
#else
    // --- manual staging fallback (global b128 -> ds b128) ---
    {
      const int srow = tid >> 1;           // 0..127
      const int scol = (tid & 1) * 16;     // 0/16
      const bf16* ga = A + (size_t)(mb + srow) * K + kb + scol;
      v8bf a0 = ((const v8bf*)ga)[0];
      v8bf a1 = ((const v8bf*)ga)[1];
      *(v8bf*)&lA[srow * 40 + scol]     = a0;
      *(v8bf*)&lA[srow * 40 + scol + 8] = a1;
      const bf16* gb = W + (size_t)(nb + srow) * K + kb + scol;
      v8bf b0 = ((const v8bf*)gb)[0];
      v8bf b1 = ((const v8bf*)gb)[1];
      *(v8bf*)&lB[srow * 40 + scol]     = b0;
      *(v8bf*)&lB[srow * 40 + scol + 8] = b1;
    }
#endif
    __syncthreads();

    // --- fragments ---
    v16bf af[2], wf[4];
#pragma unroll
    for (int mt = 0; mt < 2; mt++) {
      const bf16* p = &lA[(wm + mt * 16 + r) * 40 + hh * 8];  // A layout
      af[mt] = ld_frag16(p, p + 16);
    }
#pragma unroll
    for (int nt = 0; nt < 4; nt++) {
      const bf16* p = &lB[(wn + nt * 16 + r) * 40 + hh * 8];  // B layout
      wf[nt] = ld_frag16(p, p + 16);
    }
#pragma unroll
    for (int mt = 0; mt < 2; mt++)
#pragma unroll
      for (int nt = 0; nt < 4; nt++)
        acc[mt][nt] = WMMA_BF16(af[mt], wf[nt], acc[mt][nt]);
    __syncthreads();
  }

  // --- epilogue: bias add + store (C layout: lane = column, VGPR = row) ---
#pragma unroll
  for (int mt = 0; mt < 2; mt++)
#pragma unroll
    for (int nt = 0; nt < 4; nt++) {
      int nn = nb + wn + nt * 16 + r;
      float bv = bias[nn];
#pragma unroll
      for (int i = 0; i < 8; i++) {
        int mm = mb + wm + mt * 16 + hh * 8 + i;
        float v = acc[mt][nt][i] + bv;
        if (OUT_F32) Cf[(size_t)mm * Nn + nn] = v;
        else         Cb[(size_t)mm * Nn + nn] = (bf16)v;
      }
    }
}

// ---------------------------------------------------------------------------
// RoPE applied in place to q and k inside the packed qkv bf16 buffer.
// qkv layout: (b, n, 3*1024); head_dim=64 -> 32 (cos,sin) pairs per head.
// ---------------------------------------------------------------------------
__global__ void rope_kernel(bf16* __restrict__ qkv) {
  int idx = blockIdx.x * blockDim.x + threadIdx.x;
  int jj  = idx & 31;  idx >>= 5;   // pair index 0..31
  int h   = idx & 15;  idx >>= 4;   // head
  int sel = idx & 1;   idx >>= 1;   // 0=q, 1=k
  int n   = idx & 1023; idx >>= 10; // token
  int b   = idx;                    // batch
  int gh = n >> 5, gw = n & 31;
  int t  = jj >> 1;
  float theta = __powf(10000.f, -(float)(2 * t) * (1.f / 32.f));
  float pos   = (jj & 1) ? (float)gw : (float)gh;
  float sn, cs;
  __sincosf(pos * theta, &sn, &cs);
  bf16* p = qkv + ((size_t)(b * 1024 + n)) * 3072 + sel * 1024 + h * 64 + 2 * jj;
  float xr = (float)p[0], xi = (float)p[1];
  p[0] = (bf16)(xr * cs - xi * sn);
  p[1] = (bf16)(xr * sn + xi * cs);
}

// ---------------------------------------------------------------------------
// Flash-style attention. grid = (B*H, N/128), 256 thr = 8 waves, 16 q/wave.
// Computes S^T = K*Q^T per tile so softmax stats stay per-lane; P^T registers
// feed directly as the B operand of O^T += V^T * P^T (one cross-half shfl).
// K tile staged by TDM (pad 4 DWORDs per 32-DWORD row -> stride 72); V tile
// transposed into LDS by the threads (TDM cannot transpose 16-bit data).
// ---------------------------------------------------------------------------
__global__ __launch_bounds__(256) void attn_kernel(const bf16* __restrict__ qkv,
                                                   bf16* __restrict__ o) {
  __shared__ bf16 lK[32 * 72];   // K block, row-major (key, d), padded
  __shared__ bf16 lVT[64 * 40];  // V block transposed (d, key), padded

  const int tid  = threadIdx.x;
  const int lane = tid & 31;
  const int wv   = tid >> 5;
  const int r    = lane & 15;
  const int hh   = lane >> 4;
  const int bh = blockIdx.x;               // 0..127
  const int b  = bh >> 4, h = bh & 15;
  const int q0 = blockIdx.y * 128 + wv * 16;
  const float scale = 0.125f;              // 64^-0.5

  // Q B-fragments (d = 0..31, 32..63), loaded once, straight from global
  const bf16* qptr = qkv + ((size_t)(b * 1024 + q0 + r)) * 3072 + h * 64;
  v16bf qf0 = ld_frag16(qptr + hh * 8,      qptr + hh * 8 + 16);
  v16bf qf1 = ld_frag16(qptr + 32 + hh * 8, qptr + 32 + hh * 8 + 16);

  v8f acc[4] = {};
  float mrun = -1e30f, lrun = 0.f;

  const int skey = tid >> 3;          // staging: key 0..31
  const int sc   = (tid & 7) * 8;     // staging: d chunk
  const bf16* kg = qkv + ((size_t)(b * 1024)) * 3072 + 1024 + h * 64;
  const bf16* vg = kg + 1024;

  for (int kb = 0; kb < 1024; kb += 32) {
#if USE_TDM
    // --- K tile via TDM (issued first so the DMA overlaps the V staging) ---
    if (wv == 0)
      tdm_load_2d((unsigned)(size_t)&lK[0], kg + (size_t)kb * 3072,
                  64u, 32u, 3072ull, 4u, 3u);
#else
    {
      v8bf kv = *(const v8bf*)(kg + (size_t)(kb + skey) * 3072 + sc);
      *(v8bf*)&lK[skey * 72 + sc] = kv;
    }
#endif
    // --- stage V transposed into LDS ---
    v8bf vvv = *(const v8bf*)(vg + (size_t)(kb + skey) * 3072 + sc);
#pragma unroll
    for (int j = 0; j < 8; j++) lVT[(sc + j) * 40 + skey] = vvv[j];
#if USE_TDM
    if (wv == 0) __builtin_amdgcn_s_wait_tensorcnt(0);
#endif
    __syncthreads();

    // --- S^T = K * Q^T (two 16-key tiles, K-dim d=64 -> 2 wmmas each) ---
    v8f s0 = {}, s1 = {};
    {
      const bf16* p = &lK[(r) * 72 + hh * 8];
      s0 = WMMA_BF16(ld_frag16(p,      p + 16), qf0, s0);
      s0 = WMMA_BF16(ld_frag16(p + 32, p + 48), qf1, s0);
      p = &lK[(16 + r) * 72 + hh * 8];
      s1 = WMMA_BF16(ld_frag16(p,      p + 16), qf0, s1);
      s1 = WMMA_BF16(ld_frag16(p + 32, p + 48), qf1, s1);
    }

    // --- online softmax (per query = per lane, one cross-half shuffle) ---
    float mloc = -1e30f;
#pragma unroll
    for (int i = 0; i < 8; i++) {
      s0[i] *= scale; s1[i] *= scale;
      mloc = fmaxf(mloc, fmaxf(s0[i], s1[i]));
    }
    mloc = fmaxf(mloc, __shfl_xor(mloc, 16, 32));
    float mnew  = fmaxf(mrun, mloc);
    float alpha = __expf(mrun - mnew);
    float p0[8], p1[8], ls = 0.f;
#pragma unroll
    for (int i = 0; i < 8; i++) {
      p0[i] = __expf(s0[i] - mnew);
      p1[i] = __expf(s1[i] - mnew);
      ls += p0[i] + p1[i];
    }
    ls += __shfl_xor(ls, 16, 32);
    lrun = lrun * alpha + ls;
    mrun = mnew;
#pragma unroll
    for (int dt = 0; dt < 4; dt++)
#pragma unroll
      for (int i = 0; i < 8; i++) acc[dt][i] *= alpha;

    // --- pack P^T into the WMMA B layout (keys 0..15 | 16..31 per half) ---
    v16bf pf;
#pragma unroll
    for (int i = 0; i < 8; i++) {
      float o0 = __shfl_xor(p0[i], 16, 32);
      float o1 = __shfl_xor(p1[i], 16, 32);
      pf[i]     = (bf16)(hh ? o1 : p0[i]);   // elems 0..7
      pf[i + 8] = (bf16)(hh ? p1[i] : o0);   // elems 8..15
    }

    // --- O^T += V^T * P^T  (4 d-tiles, K=32 keys) ---
#pragma unroll
    for (int dt = 0; dt < 4; dt++) {
      const bf16* p = &lVT[(dt * 16 + r) * 40 + hh * 16];
      acc[dt] = WMMA_BF16(ld_frag16(p, p + 8), pf, acc[dt]);
    }
    __syncthreads();
  }

  // --- normalize + store (O^T: lane = query, VGPR = d) ---
  float inv = 1.f / lrun;
  bf16* optr = o + ((size_t)(b * 1024 + q0 + r)) * 1024 + h * 64;
#pragma unroll
  for (int dt = 0; dt < 4; dt++)
#pragma unroll
    for (int i = 0; i < 8; i++)
      optr[dt * 16 + hh * 8 + i] = (bf16)(acc[dt][i] * inv);
}

// ---------------------------------------------------------------------------
extern "C" void kernel_launch(void* const* d_in, const int* in_sizes, int n_in,
                              void* d_out, int out_size, void* d_ws, size_t ws_size,
                              hipStream_t stream) {
  const float* x      = (const float*)d_in[0];   // (8,1024,1024)
  const float* wqkv_w = (const float*)d_in[1];   // (3072,1024)
  const float* wqkv_b = (const float*)d_in[2];   // (3072,)
  const float* out_w  = (const float*)d_in[3];   // (1024,1024)
  const float* out_b  = (const float*)d_in[4];   // (1024,)
  float* out = (float*)d_out;                    // (8,1024,1024) f32

  // workspace layout (bf16), total ~88 MB
  bf16* xb  = (bf16*)d_ws;                          // 8M elems
  bf16* wqb = xb  + (size_t)8 * 1024 * 1024;        // 3M
  bf16* owb = wqb + (size_t)3 * 1024 * 1024;        // 1M
  bf16* qkv = owb + (size_t)1024 * 1024;            // 24M
  bf16* ob  = qkv + (size_t)8 * 1024 * 3072;        // 8M

  cvt_kernel<<<(8 * 1024 * 1024 / 4) / 256, 256, 0, stream>>>(x, xb, 8 * 1024 * 1024 / 4);
  cvt_kernel<<<(3 * 1024 * 1024 / 4) / 256, 256, 0, stream>>>(wqkv_w, wqb, 3 * 1024 * 1024 / 4);
  cvt_kernel<<<(1024 * 1024 / 4) / 256, 256, 0, stream>>>(out_w, owb, 1024 * 1024 / 4);

  // QKV projection: (8192 x 3072) = X(8192x1024) @ Wqkv^T
  gemm_kernel<false><<<dim3(3072 / 128, 8192 / 128), 256, 0, stream>>>(
      xb, wqb, wqkv_b, nullptr, qkv, 8192, 3072, 1024);

  // RoPE on q and k, in place (8*1024*2*16*32 rotations)
  rope_kernel<<<(8 * 1024 * 2 * 16 * 32) / 256, 256, 0, stream>>>(qkv);

  // attention: grid (B*H, N/128)
  attn_kernel<<<dim3(128, 8), 256, 0, stream>>>(qkv, ob);

  // output projection: (8192 x 1024) = O @ out_w^T + out_b, f32 result
  gemm_kernel<true><<<dim3(1024 / 128, 8192 / 128), 256, 0, stream>>>(
      ob, owb, out_b, out, nullptr, 8192, 1024, 1024);
}
